// GRUResidualModel_18073222381574
// MI455X (gfx1250) — matrix operand order repaired and checked by reading, not compiled, not run on hardware
//
#include <hip/hip_runtime.h>

// ---------------- problem constants ----------------
#define BB 64
#define TT 512
#define II 512
#define HH 1024
#define GG 3072   // 3*H
#define OO 128
#define KSEG 512  // compile-time K segment (II, and HH/2 for split-K)
#define JWG 32    // hidden columns per workgroup in the recurrence
#define NWGS (HH / JWG)   // 32 persistent workgroups
#define NQ 4      // n-tiles per wave in the big gx0 GEMM

typedef __bf16 bf16_t;
typedef bf16_t v16bf __attribute__((ext_vector_type(16)));
typedef bf16_t v8bf  __attribute__((ext_vector_type(8)));
typedef float  v8f   __attribute__((ext_vector_type(8)));

// ---------------- grid-wide barrier ----------------
__device__ __forceinline__ void gsync(unsigned* bar, unsigned nwg) {
    __syncthreads();
    if (threadIdx.x == 0) {
        __threadfence();
        volatile unsigned* vgen = bar + 1;
        unsigned g = *vgen;
        unsigned prev = atomicAdd(bar, 1u);
        if (prev == nwg - 1u) {
            *bar = 0u;
            __threadfence();
            atomicAdd(bar + 1, 1u);
        } else {
            while (*vgen == g) { __builtin_amdgcn_s_sleep(2); }
        }
    }
    __syncthreads();
}

// Fragment layouts (wave32, v_wmma_f32_16x16x32_bf16):
//   A 16x32: K = (e%8) + 8*(lane/16) + 16*(e/8), M = lane%16  -> two 16B runs/lane
//   B 32x16: K = (e%8) + 8*(lane/16) + 16*(e/8), N = lane%16  -> two 16B runs/lane
//   C 16x16: M = v + 8*(lane/16),   N = lane%16
__device__ __forceinline__ v16bf mk_frag(v8bf lo, v8bf hi) {
    v16bf a;
#pragma unroll
    for (int e = 0; e < 8; ++e) { a[e] = lo[e]; a[8 + e] = hi[e]; }
    return a;
}

template<int NB> struct FragGroup { v16bf a; v16bf b[NB]; };

template<int NB>
__device__ __forceinline__ FragGroup<NB> ld_group(const bf16_t* __restrict__ ap,
                                                  const bf16_t* const (&wp)[NB], int k0) {
    FragGroup<NB> f;
    f.a = mk_frag(*(const v8bf*)(ap + k0), *(const v8bf*)(ap + k0 + 16));
#pragma unroll
    for (int i = 0; i < NB; ++i)
        f.b[i] = mk_frag(*(const v8bf*)(wp[i] + k0), *(const v8bf*)(wp[i] + k0 + 16));
    return f;
}

template<int NB>
__device__ __forceinline__ void do_wmma(const FragGroup<NB>& f, v8f (&acc)[NB]) {
#pragma unroll
    for (int i = 0; i < NB; ++i)
        acc[i] = __builtin_amdgcn_wmma_f32_16x16x32_bf16(false, f.a, false, f.b[i],
                                                         (short)0, acc[i], false, false);
}

// Ping-pong pipelined K=512 accumulation: loads of group k+1 stay above WMMAs of group k.
template<int NB>
__device__ __forceinline__ void wmma_core512(const bf16_t* __restrict__ ap,
                                             const bf16_t* const (&wp)[NB], v8f (&acc)[NB]) {
    FragGroup<NB> p = ld_group<NB>(ap, wp, 0);
#pragma unroll
    for (int k0 = 0; k0 < KSEG; k0 += 64) {
        FragGroup<NB> q = ld_group<NB>(ap, wp, k0 + 32);
        __builtin_amdgcn_sched_barrier(0);
        do_wmma<NB>(p, acc);
        if (k0 + 64 < KSEG) p = ld_group<NB>(ap, wp, k0 + 64);
        __builtin_amdgcn_sched_barrier(0);
        do_wmma<NB>(q, acc);
    }
}

// ---------------- kernels ----------------
__global__ void cvt_f32_bf16(const float* __restrict__ src, bf16_t* __restrict__ dst, int n) {
    for (int i = blockIdx.x * blockDim.x + threadIdx.x; i < n; i += gridDim.x * blockDim.x)
        dst[i] = (bf16_t)src[i];
}

__global__ void init_state(float* h0, float* h2, bf16_t* h0b0, bf16_t* h0b1,
                           bf16_t* h2b0, bf16_t* h2b1, unsigned* bar) {
    int i = blockIdx.x * blockDim.x + threadIdx.x;
    for (int idx = i; idx < BB * HH; idx += gridDim.x * blockDim.x) {
        h0[idx] = 0.f; h2[idx] = 0.f;
        h0b0[idx] = (bf16_t)0.f; h0b1[idx] = (bf16_t)0.f;
        h2b0[idx] = (bf16_t)0.f; h2b1[idx] = (bf16_t)0.f;
    }
    if (i == 0) { bar[0] = 0u; bar[1] = 0u; }
}

// gx0 = x_bf16 [B*T][I] @ w_ih0_bf16[G][I]^T -> gx0 f32 (streamed, NT stores)
__global__ void __launch_bounds__(128)
gemm_gx0(const bf16_t* __restrict__ xb, const bf16_t* __restrict__ wih0,
         float* __restrict__ gx0) {
    const int lane = threadIdx.x & 31;
    const int wave = threadIdx.x >> 5;     // m-tile
    const int lh = lane >> 4, ln = lane & 15;
    const int NQUADS = GG / (16 * NQ);     // 48
    const int nItems = (BB * TT / 64) * NQUADS;
    for (int item = blockIdx.x; item < nItems; item += gridDim.x) {
        int mq = item / NQUADS;
        int nq = item % NQUADS;
        const int mbase = wave * 16;
        const int nbase0 = nq * (16 * NQ);
        const bf16_t* ap = xb + (size_t)mq * 64 * II + (size_t)(mbase + ln) * II + lh * 8;
        const bf16_t* wp[NQ];
#pragma unroll
        for (int q = 0; q < NQ; ++q)
            wp[q] = wih0 + (size_t)(nbase0 + q * 16 + ln) * II + lh * 8;
        v8f acc[NQ] = {};
        wmma_core512<NQ>(ap, wp, acc);
        float* cbase = gx0 + (size_t)mq * 64 * GG;
#pragma unroll
        for (int q = 0; q < NQ; ++q)
#pragma unroll
            for (int v = 0; v < 8; ++v) {
                int m = mbase + v + 8 * lh;
                __builtin_nontemporal_store(acc[q][v],
                    cbase + (size_t)m * GG + nbase0 + q * 16 + ln);
            }
    }
}

__device__ __forceinline__ float sigmoidf_(float x) { return 1.0f / (1.0f + __expf(-x)); }

// persistent fused 2-layer GRU: gates fused into GEMM WGs, ONE grid sync per step
__global__ void __launch_bounds__(256)
gru_seq(const float* __restrict__ gx0,
        const bf16_t* __restrict__ whh0, const bf16_t* __restrict__ wih1,
        const bf16_t* __restrict__ whh1,
        const float* __restrict__ b_ih0, const float* __restrict__ b_hh0,
        const float* __restrict__ b_ih1, const float* __restrict__ b_hh1,
        float* __restrict__ h0, float* __restrict__ h2,
        bf16_t* __restrict__ h0b0, bf16_t* __restrict__ h0b1,
        bf16_t* __restrict__ h2b0, bf16_t* __restrict__ h2b1,
        unsigned* bar) {
    __shared__ float redA[4][6][256];   // 24 KB: split-K partials (k-half 1, or gx1)
    __shared__ float redB[8][6][256];   // 48 KB: gh1 partials from all 8 waves
    const int lane = threadIdx.x & 31;
    const int wave = threadIdx.x >> 5;
    const int mw = wave & 3;            // m-tile (batch rows)
    const int kh = wave >> 2;           // k-half
    const int lh = lane >> 4, ln = lane & 15;
    const int jbase = blockIdx.x * JWG;
    const size_t arow_off = (size_t)(mw * 16 + ln) * HH + (size_t)kh * KSEG + lh * 8;

    bf16_t* hb[2] = {h0b0, h0b1};
    bf16_t* cb[2] = {h2b0, h2b1};

    // B-row offsets for the 6 strip-tiles: i -> strip s=i>>1 (r/z/n), tile q=i&1
    size_t wro[6];
#pragma unroll
    for (int i = 0; i < 6; ++i)
        wro[i] = (size_t)((i >> 1) * HH + jbase + (i & 1) * 16 + ln) * HH
                 + (size_t)kh * KSEG + lh * 8;

    for (int t = 0; t < TT; ++t) {
        const int p = t & 1;
        // ---- P_a: gh0 strips (h0 @ w_hh0^T) + layer-0 gates -> h0/h0b[1-p]
        {
            const bf16_t* ap = hb[p] + arow_off;
            const bf16_t* wp[6];
#pragma unroll
            for (int i = 0; i < 6; ++i) wp[i] = whh0 + wro[i];
            v8f acc[6] = {};
            wmma_core512<6>(ap, wp, acc);
            __syncthreads();
            if (kh) {
#pragma unroll
                for (int i = 0; i < 6; ++i)
#pragma unroll
                    for (int v = 0; v < 8; ++v) redA[mw][i][v * 32 + lane] = acc[i][v];
            }
            __syncthreads();
            if (!kh) {
#pragma unroll
                for (int q = 0; q < 2; ++q)
#pragma unroll
                    for (int v = 0; v < 8; ++v) {
                        int m = mw * 16 + v + 8 * lh;
                        int j = jbase + q * 16 + ln;
                        float ghr = acc[0 + q][v] + redA[mw][0 + q][v * 32 + lane];
                        float ghz = acc[2 + q][v] + redA[mw][2 + q][v * 32 + lane];
                        float ghn = acc[4 + q][v] + redA[mw][4 + q][v * 32 + lane];
                        const float* grow = gx0 + ((size_t)m * TT + t) * GG;
                        float r = sigmoidf_((grow[j]        + b_ih0[j])        + (ghr + b_hh0[j]));
                        float z = sigmoidf_((grow[HH + j]   + b_ih0[HH + j])   + (ghz + b_hh0[HH + j]));
                        float n = tanhf((grow[2 * HH + j] + b_ih0[2 * HH + j]) +
                                        r * (ghn + b_hh0[2 * HH + j]));
                        float hv = (1.0f - z) * n + z * h0[(size_t)m * HH + j];
                        h0[(size_t)m * HH + j] = hv;
                        hb[1 - p][(size_t)m * HH + j] = (bf16_t)hv;
                    }
            }
        }
        gsync(bar, gridDim.x);   // h1_t fully visible; ONLY barrier per step
        // ---- P_b: gh1 = h2@w_hh1^T and gx1 = h1_t@w_ih1^T + layer-1 gates
        {
            const bf16_t* a1 = cb[p] + arow_off;       // h2_{t-1}
            const bf16_t* axp = hb[1 - p] + arow_off;  // h1_t
            const bf16_t* wp1[6];
            const bf16_t* wpx[6];
#pragma unroll
            for (int i = 0; i < 6; ++i) { wp1[i] = whh1 + wro[i]; wpx[i] = wih1 + wro[i]; }
            v8f ag[6] = {};
            wmma_core512<6>(a1, wp1, ag);
            __syncthreads();
#pragma unroll
            for (int i = 0; i < 6; ++i)   // park gh1 partials (all 8 waves), free regs
#pragma unroll
                for (int v = 0; v < 8; ++v) redB[wave][i][v * 32 + lane] = ag[i][v];
            v8f ax[6] = {};
            wmma_core512<6>(axp, wpx, ax);
            __syncthreads();
            if (kh) {
#pragma unroll
                for (int i = 0; i < 6; ++i)
#pragma unroll
                    for (int v = 0; v < 8; ++v) redA[mw][i][v * 32 + lane] = ax[i][v];
            }
            __syncthreads();
            if (!kh) {
#pragma unroll
                for (int q = 0; q < 2; ++q)
#pragma unroll
                    for (int v = 0; v < 8; ++v) {
                        int m = mw * 16 + v + 8 * lh;
                        int j = jbase + q * 16 + ln;
                        int e = v * 32 + lane;
                        float gxr = ax[0 + q][v] + redA[mw][0 + q][e];
                        float gxz = ax[2 + q][v] + redA[mw][2 + q][e];
                        float gxn = ax[4 + q][v] + redA[mw][4 + q][e];
                        float ghr = redB[mw][0 + q][e] + redB[4 + mw][0 + q][e];
                        float ghz = redB[mw][2 + q][e] + redB[4 + mw][2 + q][e];
                        float ghn = redB[mw][4 + q][e] + redB[4 + mw][4 + q][e];
                        float r = sigmoidf_((gxr + b_ih1[j])        + (ghr + b_hh1[j]));
                        float z = sigmoidf_((gxz + b_ih1[HH + j])   + (ghz + b_hh1[HH + j]));
                        float n = tanhf((gxn + b_ih1[2 * HH + j]) +
                                        r * (ghn + b_hh1[2 * HH + j]));
                        float hv = (1.0f - z) * n + z * h2[(size_t)m * HH + j];
                        h2[(size_t)m * HH + j] = hv;
                        cb[1 - p][(size_t)m * HH + j] = (bf16_t)hv;
                    }
            }
        }
        // no barrier needed here: P_b(t) and P_a(t+1) touch disjoint buffers;
        // LDS hazards are covered by the __syncthreads inside P_a(t+1).
    }
}

// last-timestep projection + residual + batchnorm -> vbuf[64][1024]
__global__ void finalize_bn(const float* __restrict__ x, const float* __restrict__ w_proj,
                            const float* __restrict__ b_proj, const float* __restrict__ h2,
                            const float* __restrict__ gamma, const float* __restrict__ beta,
                            const float* __restrict__ mean, const float* __restrict__ var,
                            float* __restrict__ vbuf) {
    for (int idx = blockIdx.x * blockDim.x + threadIdx.x; idx < BB * HH;
         idx += gridDim.x * blockDim.x) {
        int b = idx >> 10, j = idx & (HH - 1);
        const float* xr = x + ((size_t)b * TT + (TT - 1)) * II;
        const float* wr = w_proj + (size_t)j * II;
        float acc = b_proj[j];
        for (int i = 0; i < II; ++i) acc = fmaf(xr[i], wr[i], acc);
        float v = acc + h2[idx];
        v = (v - mean[j]) * __frsqrt_rn(var[j] + 1e-5f) * gamma[j] + beta[j];
        vbuf[idx] = v;
    }
}

// out = vbuf @ w_out^T + b_out   (64x128, tiny)
__global__ void finalize_out(const float* __restrict__ vbuf, const float* __restrict__ w_out,
                             const float* __restrict__ b_out, float* __restrict__ out) {
    for (int idx = blockIdx.x * blockDim.x + threadIdx.x; idx < BB * OO;
         idx += gridDim.x * blockDim.x) {
        int b = idx >> 7, o = idx & (OO - 1);
        const float* vr = vbuf + (size_t)b * HH;
        const float* wr = w_out + (size_t)o * HH;
        float acc = b_out[o];
        for (int j = 0; j < HH; ++j) acc = fmaf(vr[j], wr[j], acc);
        out[idx] = acc;
    }
}

// ---------------- launcher ----------------
extern "C" void kernel_launch(void* const* d_in, const int* in_sizes, int n_in,
                              void* d_out, int out_size, void* d_ws, size_t ws_size,
                              hipStream_t stream) {
    (void)in_sizes; (void)n_in; (void)out_size; (void)ws_size;
    const float* x      = (const float*)d_in[0];
    const float* w_proj = (const float*)d_in[1];
    const float* b_proj = (const float*)d_in[2];
    const float* w_ih0  = (const float*)d_in[3];
    const float* w_hh0  = (const float*)d_in[4];
    const float* b_ih0  = (const float*)d_in[5];
    const float* b_hh0  = (const float*)d_in[6];
    const float* w_ih1  = (const float*)d_in[7];
    const float* w_hh1  = (const float*)d_in[8];
    const float* b_ih1  = (const float*)d_in[9];
    const float* b_hh1  = (const float*)d_in[10];
    const float* bn_g   = (const float*)d_in[11];
    const float* bn_b   = (const float*)d_in[12];
    const float* bn_m   = (const float*)d_in[13];
    const float* bn_v   = (const float*)d_in[14];
    const float* w_out  = (const float*)d_in[15];
    const float* b_out  = (const float*)d_in[16];
    float* out = (float*)d_out;

    char* ws = (char*)d_ws;
    size_t off = 0;
    auto take = [&](size_t bytes) { size_t o = off; off += (bytes + 255) & ~(size_t)255; return o; };

    unsigned* bar   = (unsigned*)(ws + take(256));
    bf16_t* xb      = (bf16_t*)(ws + take((size_t)BB * TT * II * 2));
    bf16_t* wih0b   = (bf16_t*)(ws + take((size_t)GG * II * 2));
    bf16_t* whh0b   = (bf16_t*)(ws + take((size_t)GG * HH * 2));
    bf16_t* wih1b   = (bf16_t*)(ws + take((size_t)GG * HH * 2));
    bf16_t* whh1b   = (bf16_t*)(ws + take((size_t)GG * HH * 2));
    float*  gx0     = (float*)(ws + take((size_t)BB * TT * GG * 4));
    float*  h0      = (float*)(ws + take((size_t)BB * HH * 4));
    float*  h2      = (float*)(ws + take((size_t)BB * HH * 4));
    bf16_t* h0b0    = (bf16_t*)(ws + take((size_t)BB * HH * 2));
    bf16_t* h0b1    = (bf16_t*)(ws + take((size_t)BB * HH * 2));
    bf16_t* h2b0    = (bf16_t*)(ws + take((size_t)BB * HH * 2));
    bf16_t* h2b1    = (bf16_t*)(ws + take((size_t)BB * HH * 2));
    float*  vbuf    = (float*)(ws + take((size_t)BB * HH * 4));

    // 1) precision conversion (bf16 weights stay resident in 192MB L2)
    cvt_f32_bf16<<<2048, 256, 0, stream>>>(x,     xb,    BB * TT * II);
    cvt_f32_bf16<<<512,  256, 0, stream>>>(w_ih0, wih0b, GG * II);
    cvt_f32_bf16<<<512,  256, 0, stream>>>(w_hh0, whh0b, GG * HH);
    cvt_f32_bf16<<<512,  256, 0, stream>>>(w_ih1, wih1b, GG * HH);
    cvt_f32_bf16<<<512,  256, 0, stream>>>(w_hh1, whh1b, GG * HH);
    init_state<<<512, 256, 0, stream>>>(h0, h2, h0b0, h0b1, h2b0, h2b1, bar);

    // 2) T-parallel GEMM: gx0 = x @ w_ih0^T (bf16 WMMA, NT stores)
    gemm_gx0<<<4096, 128, 0, stream>>>(xb, wih0b, gx0);

    // 3) persistent fused 2-layer GRU (512 steps, ONE grid sync per step)
    gru_seq<<<NWGS, 256, 0, stream>>>(gx0, whh0b, wih1b, whh1b,
                                      b_ih0, b_hh0, b_ih1, b_hh1,
                                      h0, h2, h0b0, h0b1, h2b0, h2b1, bar);

    // 4) epilogue: last-step projection + residual + BN, then tiny output GEMM
    finalize_bn<<<256, 256, 0, stream>>>(x, w_proj, b_proj, h2, bn_g, bn_b, bn_m, bn_v, vbuf);
    finalize_out<<<32, 256, 0, stream>>>(vbuf, w_out, b_out, out);
}
